// MultiHeadAttention_76897094467762
// MI455X (gfx1250) — matrix operand order repaired and checked by reading, not compiled, not run on hardware
//
#include <hip/hip_runtime.h>

// MI455X / gfx1250. All matmuls: v_wmma_f32_16x16x32_f16 with split-f16
// operands (hi = f16(x), lo = f16(x - hi); D += Ah*Bh + Ah*Bl + Al*Bh)
// giving ~fp32 accuracy at f16 WMMA rates. All tensors are pre-split ONCE
// (or produced pre-split by the previous kernel's epilogue) so GEMM inner
// loops are pure b128 loads + WMMA.
// Workspace peak (with overlays): ~101 MB.

typedef __attribute__((ext_vector_type(16))) _Float16 v16h;
typedef __attribute__((ext_vector_type(8)))  _Float16 v8h;
typedef __attribute__((ext_vector_type(8)))  float    v8f;

#define DMODEL 512
#define NHEAD  8
#define DHEAD  64
#define DFF    2048

__device__ __forceinline__ void split_f16(float x, _Float16& hi, _Float16& lo) {
  _Float16 h = (_Float16)x;
  hi = h;
  lo = (_Float16)(x - (float)h);
}

__device__ __forceinline__ v16h cat8(v8h a, v8h b) {
  return __builtin_shufflevector(a, b, 0,1,2,3,4,5,6,7,8,9,10,11,12,13,14,15);
}

// ---- fragment loaders from pre-split f16 memory -----------------------------
// A fragment 16(M) x 32(K), source row-major halves. ISA layout: lane m=L&15,
// half=L>>4; elems 0..7 -> K = half*8+0..7, elems 8..15 -> K = 16+half*8+0..7.
__device__ __forceinline__ v16h load_a16(const _Float16* __restrict__ A, int lda, int lane) {
  const _Float16* p = A + (size_t)(lane & 15) * lda + ((lane >> 4) * 8);
  return cat8(*(const v8h*)p, *(const v8h*)(p + 16));
}
// B fragment 32(K) x 16(N) from [N][K] halves (K contiguous). lane n=L&15,
// half=L>>4; elems 0..7 -> K = half*8+0..7, elems 8..15 -> K = 16+half*8+0..7.
__device__ __forceinline__ v16h load_b16(const _Float16* __restrict__ B, int ldb, int lane) {
  const _Float16* p = B + (size_t)(lane & 15) * ldb + ((lane >> 4) * 8);
  return cat8(*(const v8h*)p, *(const v8h*)(p + 16));
}
// A fragment with on-the-fly split from f32 (used only for P from LDS).
__device__ __forceinline__ void load_a_frag_f32(const float* __restrict__ A, int lda, int lane,
                                                v16h& hi, v16h& lo) {
  const int m = lane & 15, kb = (lane >> 4) * 8;
  const float* p0 = A + (size_t)m * lda + kb;
  const float* p1 = p0 + 16;
#pragma unroll
  for (int e = 0; e < 8; ++e) {
    _Float16 h, l;
    split_f16(p0[e], h, l); hi[e] = h;     lo[e] = l;
    split_f16(p1[e], h, l); hi[8 + e] = h; lo[8 + e] = l;
  }
}

__device__ __forceinline__ v8f wmma3(v16h ah, v16h al, v16h bh, v16h bl, v8f c) {
  c = __builtin_amdgcn_wmma_f32_16x16x32_f16(false, ah, false, bh, (short)0, c, false, false);
  c = __builtin_amdgcn_wmma_f32_16x16x32_f16(false, ah, false, bl, (short)0, c, false, false);
  c = __builtin_amdgcn_wmma_f32_16x16x32_f16(false, al, false, bh, (short)0, c, false, false);
  return c;
}

// ---------------------------------------------------------------------------
// split_pack: f32[n] -> hi f16[n], lo f16[n]   (n % 8 == 0)
// ---------------------------------------------------------------------------
__global__ __launch_bounds__(256) void split_pack(const float* __restrict__ X,
                                                  _Float16* __restrict__ H,
                                                  _Float16* __restrict__ L,
                                                  long long n) {
  const long long i = ((long long)blockIdx.x * blockDim.x + threadIdx.x) * 8;
  if (i >= n) return;
  v8h h, l;
#pragma unroll
  for (int e = 0; e < 8; ++e) {
    _Float16 hh, ll;
    split_f16(X[i + e], hh, ll);
    h[e] = hh; l[e] = ll;
  }
  *(v8h*)(H + i) = h;
  *(v8h*)(L + i) = l;
}

// ---------------------------------------------------------------------------
// GEMM: C[M,N] = A[M,K] @ W[N,K]^T (+bias)(+relu). A and W pre-split f16.
// One wave computes a 32(M) x 64(N) tile (8 accumulators, 24 WMMA / k-chunk).
// Output: f32 (Cf) and/or split-f16 (Ch/Cl), optionally transposed [N][M].
// ---------------------------------------------------------------------------
__global__ __launch_bounds__(256) void gemm_wmma_split(
    const _Float16* __restrict__ Ah, const _Float16* __restrict__ Al,
    const _Float16* __restrict__ Wh, const _Float16* __restrict__ Wl,
    float* __restrict__ Cf, _Float16* __restrict__ Ch, _Float16* __restrict__ Cl,
    const float* __restrict__ bias,
    int M, int N, int K, int relu, int transC) {
  const int lane = threadIdx.x & 31;
  const int wave = blockIdx.x * (blockDim.x >> 5) + (threadIdx.x >> 5);
  const int ntiles = N >> 6;
  const int mt = wave / ntiles;
  const int n0 = (wave % ntiles) << 6;
  if (mt * 32 >= M) return;  // wave-uniform

  const size_t aoff = (size_t)(lane & 15) * K + ((lane >> 4) * 8);
  const size_t boff = (size_t)(lane & 15) * K + ((lane >> 4) * 8);
  const _Float16* pAh0 = Ah + (size_t)mt * 32 * K + aoff;
  const _Float16* pAl0 = Al + (size_t)mt * 32 * K + aoff;
  const _Float16* pAh1 = pAh0 + (size_t)16 * K;
  const _Float16* pAl1 = pAl0 + (size_t)16 * K;
  const _Float16* pWh  = Wh + (size_t)n0 * K + boff;
  const _Float16* pWl  = Wl + (size_t)n0 * K + boff;

  v8f acc[2][4] = {};
  for (int k0 = 0; k0 < K; k0 += 32) {
    const v16h ah0 = cat8(*(const v8h*)(pAh0 + k0), *(const v8h*)(pAh0 + k0 + 16));
    const v16h al0 = cat8(*(const v8h*)(pAl0 + k0), *(const v8h*)(pAl0 + k0 + 16));
    const v16h ah1 = cat8(*(const v8h*)(pAh1 + k0), *(const v8h*)(pAh1 + k0 + 16));
    const v16h al1 = cat8(*(const v8h*)(pAl1 + k0), *(const v8h*)(pAl1 + k0 + 16));
#pragma unroll
    for (int t = 0; t < 4; ++t) {
      const _Float16* qh = pWh + (size_t)t * 16 * K + k0;
      const _Float16* ql = pWl + (size_t)t * 16 * K + k0;
      const v16h bh = cat8(*(const v8h*)qh, *(const v8h*)(qh + 16));
      const v16h bl = cat8(*(const v8h*)ql, *(const v8h*)(ql + 16));
      acc[0][t] = wmma3(ah0, al0, bh, bl, acc[0][t]);
      acc[1][t] = wmma3(ah1, al1, bh, bl, acc[1][t]);
    }
  }

  const int nlane = lane & 15;
  const int halfo = (lane >> 4) * 8;
#pragma unroll
  for (int ms = 0; ms < 2; ++ms) {
    const int mbase = mt * 32 + ms * 16 + halfo;  // C: M = vgpr + half*8
#pragma unroll
    for (int t = 0; t < 4; ++t) {
      const int col = n0 + t * 16 + nlane;
      const float bb = bias ? bias[col] : 0.0f;
      float vv[8];
#pragma unroll
      for (int r = 0; r < 8; ++r) {
        float v = acc[ms][t][r] + bb;
        if (relu) v = fmaxf(v, 0.0f);
        vv[r] = v;
      }
      if (Cf) {
#pragma unroll
        for (int r = 0; r < 8; ++r) Cf[(size_t)(mbase + r) * N + col] = vv[r];
      }
      if (Ch) {
        if (transC) {  // [N][M] halves; 8 contiguous rows -> b128 stores
          v8h hv, lv;
#pragma unroll
          for (int r = 0; r < 8; ++r) { _Float16 h2, l2; split_f16(vv[r], h2, l2); hv[r] = h2; lv[r] = l2; }
          *(v8h*)(Ch + (size_t)col * M + mbase) = hv;
          *(v8h*)(Cl + (size_t)col * M + mbase) = lv;
        } else {
#pragma unroll
          for (int r = 0; r < 8; ++r) {
            _Float16 h2, l2; split_f16(vv[r], h2, l2);
            Ch[(size_t)(mbase + r) * N + col] = h2;
            Cl[(size_t)(mbase + r) * N + col] = l2;
          }
        }
      }
    }
  }
}

// ---------------------------------------------------------------------------
// Attention: one wave per (16 query rows, head). Q/K pre-split row-major,
// V pre-split TRANSPOSED [dh][rows] so P@V B-frags are K-contiguous.
// Scores -> 64KB LDS row buffer, exact softmax, P@V; O emitted pre-split.
// All lengths % 64 == 0 -> no masking (packed layout == reference mask).
// ---------------------------------------------------------------------------
__global__ __launch_bounds__(32) void attn_wmma(
    const _Float16* __restrict__ Qh, const _Float16* __restrict__ Ql,
    const _Float16* __restrict__ Kh, const _Float16* __restrict__ Kl,
    const _Float16* __restrict__ Vth, const _Float16* __restrict__ Vtl,
    const int* __restrict__ lenx, const int* __restrict__ lenm,
    _Float16* __restrict__ Oh, _Float16* __restrict__ Ol,
    int B, int Mm) {
  __shared__ float sS[16 * 1024];
  const int lane = threadIdx.x;
  const int h    = blockIdx.y;
  const int row0 = blockIdx.x * 16;

  int mstart = 0, mlen = 0;
  {
    int ax = 0, am = 0;
    for (int i = 0; i < B; ++i) {
      const int a = lenx[i], m = lenm[i];
      if (row0 >= ax && row0 < ax + a) { mstart = am; mlen = m; }
      ax += a; am += m;
    }
  }

  v16h qh[2], ql[2];
#pragma unroll
  for (int c = 0; c < 2; ++c) {
    qh[c] = load_a16(Qh + (size_t)row0 * DMODEL + h * DHEAD + c * 32, DMODEL, lane);
    ql[c] = load_a16(Ql + (size_t)row0 * DMODEL + h * DHEAD + c * 32, DMODEL, lane);
  }

  // ---- S[16, mlen] = (Q @ K^T) / 8 ----
  for (int y0 = 0; y0 < mlen; y0 += 16) {
    v8f s = {};
#pragma unroll
    for (int c = 0; c < 2; ++c) {
      const size_t ko = (size_t)(mstart + y0) * DMODEL + h * DHEAD + c * 32;
      const v16h kh = load_b16(Kh + ko, DMODEL, lane);
      const v16h kl = load_b16(Kl + ko, DMODEL, lane);
      s = wmma3(qh[c], ql[c], kh, kl, s);
    }
    const int rr = (lane >> 4) * 8;
    const int cc = y0 + (lane & 15);
#pragma unroll
    for (int r = 0; r < 8; ++r) sS[(rr + r) * 1024 + cc] = s[r] * 0.125f;
  }
  __syncthreads();

  // ---- exact softmax per row (2 lanes per row) ----
  {
    const int r = lane & 15;
    const int p = lane >> 4;
    float* rowp = &sS[r * 1024];
    float mx = -3.0e38f;
    for (int c2 = p; c2 < mlen; c2 += 2) mx = fmaxf(mx, rowp[c2]);
    mx = fmaxf(mx, __shfl_xor(mx, 16, 32));
    float sum = 0.0f;
    for (int c2 = p; c2 < mlen; c2 += 2) {
      const float e = __expf(rowp[c2] - mx);
      rowp[c2] = e; sum += e;
    }
    sum += __shfl_xor(sum, 16, 32);
    const float inv = 1.0f / sum;
    for (int c2 = p; c2 < mlen; c2 += 2) rowp[c2] *= inv;
  }
  __syncthreads();

  // ---- O[16, 64] = P[16, mlen] @ V[mlen, 64] ----
  v8f oacc[4] = {};
  for (int y0 = 0; y0 < mlen; y0 += 32) {
    v16h ph, pl;
    load_a_frag_f32(&sS[y0], 1024, lane, ph, pl);
#pragma unroll
    for (int t = 0; t < 4; ++t) {
      const size_t vo = (size_t)(h * DHEAD + t * 16) * Mm + mstart + y0;
      const v16h vh = load_b16(Vth + vo, Mm, lane);  // Vt: [dh][rows], K contiguous
      const v16h vl = load_b16(Vtl + vo, Mm, lane);
      oacc[t] = wmma3(ph, pl, vh, vl, oacc[t]);
    }
  }
  const int mbase = row0 + (lane >> 4) * 8;
  const int nlane = lane & 15;
#pragma unroll
  for (int t = 0; t < 4; ++t)
#pragma unroll
    for (int r = 0; r < 8; ++r) {
      _Float16 h2, l2; split_f16(oacc[t][r], h2, l2);
      Oh[(size_t)(mbase + r) * DMODEL + h * DHEAD + t * 16 + nlane] = h2;
      Ol[(size_t)(mbase + r) * DMODEL + h * DHEAD + t * 16 + nlane] = l2;
    }
}

// ---------------------------------------------------------------------------
// Out = LayerNorm(X + Y) * g + b. X from f32 OR split pair; Out to f32
// and/or split pair. One wave per row (D = 512).
// ---------------------------------------------------------------------------
__global__ __launch_bounds__(256) void ln_residual(
    const float* __restrict__ Xf, const _Float16* __restrict__ Xh, const _Float16* __restrict__ Xl,
    const float* __restrict__ Y, const float* __restrict__ g, const float* __restrict__ b,
    float* __restrict__ Outf, _Float16* __restrict__ Outh, _Float16* __restrict__ Outl, int M) {
  const int row = blockIdx.x * (blockDim.x >> 5) + (threadIdx.x >> 5);
  if (row >= M) return;
  const int lane = threadIdx.x & 31;
  float v[16];
  float sum = 0.0f;
#pragma unroll
  for (int i = 0; i < 16; ++i) {
    const size_t idx = (size_t)row * DMODEL + lane + i * 32;
    const float xv = Xf ? Xf[idx] : ((float)Xh[idx] + (float)Xl[idx]);
    v[i] = xv + Y[idx];
    sum += v[i];
  }
#pragma unroll
  for (int o = 16; o > 0; o >>= 1) sum += __shfl_xor(sum, o, 32);
  const float mu = sum * (1.0f / DMODEL);
  float var = 0.0f;
#pragma unroll
  for (int i = 0; i < 16; ++i) { const float d = v[i] - mu; var += d * d; }
#pragma unroll
  for (int o = 16; o > 0; o >>= 1) var += __shfl_xor(var, o, 32);
  const float rs = rsqrtf(var * (1.0f / DMODEL) + 1e-5f);
#pragma unroll
  for (int i = 0; i < 16; ++i) {
    const int c = lane + i * 32;
    const size_t idx = (size_t)row * DMODEL + c;
    const float o = (v[i] - mu) * rs * g[c] + b[c];
    if (Outf) Outf[idx] = o;
    if (Outh) { _Float16 h2, l2; split_f16(o, h2, l2); Outh[idx] = h2; Outl[idx] = l2; }
  }
}

extern "C" void kernel_launch(void* const* d_in, const int* in_sizes, int n_in,
                              void* d_out, int out_size, void* d_ws, size_t ws_size,
                              hipStream_t stream) {
  const float* x    = (const float*)d_in[0];
  const float* mem  = (const float*)d_in[1];
  const int*   lx   = (const int*)d_in[2];
  const int*   lm   = (const int*)d_in[3];
  const float* Wq   = (const float*)d_in[4];
  const float* Wk   = (const float*)d_in[5];
  const float* Wv   = (const float*)d_in[6];
  const float* Wo   = (const float*)d_in[7];
  const float* W1   = (const float*)d_in[8];
  const float* b1   = (const float*)d_in[9];
  const float* W2   = (const float*)d_in[10];
  const float* b2   = (const float*)d_in[11];
  const float* ln1g = (const float*)d_in[12];
  const float* ln1b = (const float*)d_in[13];
  const float* ln2g = (const float*)d_in[14];
  const float* ln2b = (const float*)d_in[15];

  const int Mx = in_sizes[0] / DMODEL;  // 6336
  const int Mm = in_sizes[1] / DMODEL;  // 5952
  const int B  = in_sizes[2];           // 8

  const size_t nx  = (size_t)Mx * DMODEL;
  const size_t nm  = (size_t)Mm * DMODEL;
  const size_t nW  = (size_t)DMODEL * DMODEL;
  const size_t nW1 = (size_t)DFF * DMODEL;
  const size_t nFF = (size_t)Mx * DFF;

  // Byte offsets into ws. A split pair of n elements occupies 4n bytes
  // (hi f16[n] then lo f16[n]); an f32 tensor of n elements occupies 4n bytes.
  // Overlays (sequential phases): O over xs; FF1 over xs/ms/Q/K/Vt; FF2 over ATT.
  // Peak ~= 8nx+12nm + 4nx + 4nx + 4*4nW + 2*4nW1 bytes ~= 101 MB.
  const size_t o_xs  = 0;
  const size_t o_ms  = o_xs + 4 * nx;
  const size_t o_Q   = o_ms + 4 * nm;
  const size_t o_K   = o_Q + 4 * nx;
  const size_t o_Vt  = o_K + 4 * nm;
  const size_t endA  = o_Vt + 4 * nm;   // 8nx + 12nm
  const size_t o_O   = 0;               // over xs (dead)
  const size_t o_FF1 = 0;               // over region A (dead); 4nFF <= endA
  const size_t o_ATT = endA;
  const size_t o_FF2 = o_ATT;           // over ATT (dead)
  const size_t o_H1  = o_ATT + 4 * nx;
  const size_t o_Wq  = o_H1 + 4 * nx;
  const size_t o_Wk  = o_Wq + 4 * nW;
  const size_t o_Wv  = o_Wk + 4 * nW;
  const size_t o_Wo  = o_Wv + 4 * nW;
  const size_t o_W1  = o_Wo + 4 * nW;
  const size_t o_W2  = o_W1 + 4 * nW1;

  char* base = (char*)d_ws;
  auto Hp = [&](size_t off) { return (_Float16*)(base + off); };
  auto Fp = [&](size_t off) { return (float*)(base + off); };

  auto conv = [&](const float* src, size_t off, size_t n) {
    const int blocks = (int)((n / 8 + 255) / 256);
    split_pack<<<blocks, 256, 0, stream>>>(src, Hp(off), Hp(off) + n, (long long)n);
  };
  auto gemm = [&](size_t oA, size_t nA, size_t oW, size_t nWe,
                  float* Cf, _Float16* Ch, _Float16* Cl, const float* bias,
                  int M, int N, int K, int relu, int transC) {
    const int tiles = (M / 32) * (N / 64);
    gemm_wmma_split<<<(tiles + 7) / 8, 256, 0, stream>>>(
        Hp(oA), Hp(oA) + nA, Hp(oW), Hp(oW) + nWe,
        Cf, Ch, Cl, bias, M, N, K, relu, transC);
  };

  // 1) pre-split inputs and weights (each done exactly once)
  conv(x,   o_xs, nx);
  conv(mem, o_ms, nm);
  conv(Wq, o_Wq, nW); conv(Wk, o_Wk, nW); conv(Wv, o_Wv, nW); conv(Wo, o_Wo, nW);
  conv(W1, o_W1, nW1); conv(W2, o_W2, nW1);

  // 2) projections (outputs pre-split; V transposed [dh][rows])
  gemm(o_xs, nx, o_Wq, nW, nullptr, Hp(o_Q),  Hp(o_Q) + nx,  nullptr, Mx, DMODEL, DMODEL, 0, 0);
  gemm(o_ms, nm, o_Wk, nW, nullptr, Hp(o_K),  Hp(o_K) + nm,  nullptr, Mm, DMODEL, DMODEL, 0, 0);
  gemm(o_ms, nm, o_Wv, nW, nullptr, Hp(o_Vt), Hp(o_Vt) + nm, nullptr, Mm, DMODEL, DMODEL, 0, 1);

  // 3) packed variable-length cross-attention
  attn_wmma<<<dim3(Mx / 16, NHEAD), 32, 0, stream>>>(
      Hp(o_Q), Hp(o_Q) + nx, Hp(o_K), Hp(o_K) + nm, Hp(o_Vt), Hp(o_Vt) + nm,
      lx, lm, Hp(o_O), Hp(o_O) + nx, B, Mm);

  // 4) output projection (f32) + residual LayerNorm -> H1 (split)
  gemm(o_O, nx, o_Wo, nW, Fp(o_ATT), nullptr, nullptr, nullptr, Mx, DMODEL, DMODEL, 0, 0);
  ln_residual<<<(Mx + 7) / 8, 256, 0, stream>>>(
      x, nullptr, nullptr, Fp(o_ATT), ln1g, ln1b,
      nullptr, Hp(o_H1), Hp(o_H1) + nx, Mx);

  // 5) FFN (FF1 pre-split with fused bias+relu) + residual LayerNorm -> out
  gemm(o_H1,  nx,  o_W1, nW1, nullptr, Hp(o_FF1), Hp(o_FF1) + nFF, b1, Mx, DFF,    DMODEL, 1, 0);
  gemm(o_FF1, nFF, o_W2, nW1, Fp(o_FF2), nullptr, nullptr,         b2, Mx, DMODEL, DFF,    0, 0);
  ln_residual<<<(Mx + 7) / 8, 256, 0, stream>>>(
      nullptr, Hp(o_H1), Hp(o_H1) + nx, Fp(o_FF2), ln2g, ln2b,
      (float*)d_out, nullptr, nullptr, Mx);
}
